// TripletNNPULoss_30185030156999
// MI455X (gfx1250) — matrix operand order repaired and checked by reading, not compiled, not run on hardware
//
#include <hip/hip_runtime.h>

typedef _Float16 v16h __attribute__((ext_vector_type(16)));
typedef _Float16 v8h  __attribute__((ext_vector_type(8)));
typedef float    v8f  __attribute__((ext_vector_type(8)));
typedef float    v4f  __attribute__((ext_vector_type(4)));

#define NROWS   4096
#define CDIM    256
#define KSEL    20
#define STRIDEF 4100   // padded LDS row stride in floats (avoids bank aliasing between M=0..7 / M=8..15)

// ---------------------------------------------------------------------------
// Kernel 1: L2-normalize rows of input & target, emit f16 copies for WMMA.
// grid = 1024 blocks x 256 thr: blocks [0,512) -> input rows, [512,1024) -> target.
// One wave per row (32 lanes x 8 elems = 256).
// ---------------------------------------------------------------------------
__global__ void tnnpu_normalize(const float* __restrict__ inp,
                                const float* __restrict__ tgt,
                                _Float16* __restrict__ Af,
                                _Float16* __restrict__ Bf) {
  const int wave = threadIdx.x >> 5;
  const int lane = threadIdx.x & 31;
  const int b    = blockIdx.x;

  const float* src;
  _Float16*    dst;
  int rowBase;
  if (b < 512) { src = inp; dst = Af; rowBase = b * 8; }
  else         { src = tgt; dst = Bf; rowBase = (b - 512) * 8; }

  const int r = rowBase + wave;
  const float* p = src + (size_t)r * CDIM + lane * 8;
  v4f x0 = *(const v4f*)p;
  v4f x1 = *(const v4f*)(p + 4);

  float ss = x0[0]*x0[0] + x0[1]*x0[1] + x0[2]*x0[2] + x0[3]*x0[3]
           + x1[0]*x1[0] + x1[1]*x1[1] + x1[2]*x1[2] + x1[3]*x1[3];
  #pragma unroll
  for (int off = 16; off > 0; off >>= 1)
    ss += __shfl_xor(ss, off, 32);

  const float scale = 1.0f / fmaxf(sqrtf(ss), 1e-12f);

  v8h o;
  o[0] = (_Float16)(x0[0] * scale);
  o[1] = (_Float16)(x0[1] * scale);
  o[2] = (_Float16)(x0[2] * scale);
  o[3] = (_Float16)(x0[3] * scale);
  o[4] = (_Float16)(x1[0] * scale);
  o[5] = (_Float16)(x1[1] * scale);
  o[6] = (_Float16)(x1[2] * scale);
  o[7] = (_Float16)(x1[3] * scale);
  *(v8h*)(dst + (size_t)r * CDIM + lane * 8) = o;
}

// ---------------------------------------------------------------------------
// Kernel 2: per-16-row stripe: WMMA GEMM (dist stripe into LDS), then
// per-row top-20 min / top-20 max + nnPU loss.
// grid = 256 blocks x 256 thr (8 waves). Dynamic LDS = 16*STRIDEF*4 bytes.
// ---------------------------------------------------------------------------
__global__ void tnnpu_gemm_topk(const _Float16* __restrict__ A,
                                const _Float16* __restrict__ B,
                                float* __restrict__ lossOut) {
  extern __shared__ float stripe[];
  const int wave = threadIdx.x >> 5;
  const int lane = threadIdx.x & 31;
  const int M0   = blockIdx.x << 4;
  const int hi   = lane >> 4;     // 0: lanes 0-15, 1: lanes 16-31
  const int l15  = lane & 15;

  // ----- A fragments for this M-tile (K = 256 -> 8 chunks of 32), registers.
  // 16-bit A 16x32 layout: lane (l15) = M row; lanes<16 hold K {8h..8h+7, 16+8h..},
  // h = hi. Both chunks are contiguous 8-half (16B) loads from row-major A.
  const _Float16* Arow = A + (size_t)(M0 + l15) * CDIM;
  v16h afrag[8];
  #pragma unroll
  for (int kk = 0; kk < 8; ++kk) {
    v8h lo = *(const v8h*)(Arow + kk * 32 + hi * 8);
    v8h hh = *(const v8h*)(Arow + kk * 32 + 16 + hi * 8);
    afrag[kk] = __builtin_shufflevector(lo, hh,
        0,1,2,3,4,5,6,7,8,9,10,11,12,13,14,15);
  }

  // ----- GEMM phase: each wave covers 32 of the 256 N-tiles.
  for (int t = 0; t < 32; ++t) {
    const int nt = (t << 3) + wave;     // N-tile index 0..255
    const int N0 = nt << 4;
    // B is target_norm^T: B[k][n] = Tn[n][k] -> lane (col = N0+l15) reads 16
    // halfs of target row. 32x16 B layout: lanes<16 K={0..7,16..23}, >=16 K={8..15,24..31}.
    const _Float16* Bcol = B + (size_t)(N0 + l15) * CDIM + hi * 8;

    if (t + 1 < 32) {  // prefetch next tile's B rows (global_prefetch_b8)
      const int nn = (((t + 1) << 3) + wave) << 4;
      __builtin_prefetch(B + (size_t)(nn + l15) * CDIM, 0, 1);
    }

    v8f acc = {};
    #pragma unroll
    for (int kk = 0; kk < 8; ++kk) {
      v16h bfrag = __builtin_shufflevector(*(const v8h*)(Bcol + kk * 32),
          *(const v8h*)(Bcol + kk * 32 + 16), 0,1,2,3,4,5,6,7,8,9,10,11,12,13,14,15);
      acc = __builtin_amdgcn_wmma_f32_16x16x32_f16(
          false, afrag[kk], false, bfrag, (short)0, acc, false, false);
    }

    // C/D 16x16 f32 layout: lane col = l15, rows = hi*8 + v.
    const int rb = hi << 3;
    #pragma unroll
    for (int v = 0; v < 8; ++v)
      stripe[(rb + v) * STRIDEF + N0 + l15] = 0.5f - 0.5f * acc[v];
  }
  __syncthreads();

  // ----- Top-k phase: wave w owns rows 2w, 2w+1 of the stripe.
  for (int rr = 0; rr < 2; ++rr) {
    const int r = (wave << 1) + rr;
    const int g = M0 + r;                 // diagonal column for this row
    float* row = stripe + r * STRIDEF;
    const float dist_ap = row[g];

    // --- 20 smallest (extract-min; remember extracted entries for restore)
    float summin = 0.0f;
    float myVal = 0.0f; int myIdx = 0;
    for (int it = 0; it < KSEL; ++it) {
      float bv = __builtin_inff(); int bj = -1;
      for (int j = lane; j < NROWS; j += 32) {
        const float v = row[j];
        if (j != g && v < bv) { bv = v; bj = j; }
      }
      #pragma unroll
      for (int off = 16; off > 0; off >>= 1) {
        const float ov = __shfl_xor(bv, off, 32);
        const int   oj = __shfl_xor(bj, off, 32);
        if (ov < bv || (ov == bv && oj < bj)) { bv = ov; bj = oj; }
      }
      summin += bv;
      if (lane == it) { myVal = bv; myIdx = bj; }
      row[bj] = __builtin_inff();          // every lane stores same value -> own
    }                                      // store orders its later loads
    if (lane < KSEL) row[myIdx] = myVal;   // restore extracted minima
    __syncthreads();                       // uniform across waves; makes restore visible

    // --- 20 largest
    float summax = 0.0f;
    for (int it = 0; it < KSEL; ++it) {
      float bv = -__builtin_inff(); int bj = -1;
      for (int j = lane; j < NROWS; j += 32) {
        const float v = row[j];
        if (j != g && v > bv) { bv = v; bj = j; }
      }
      #pragma unroll
      for (int off = 16; off > 0; off >>= 1) {
        const float ov = __shfl_xor(bv, off, 32);
        const int   oj = __shfl_xor(bj, off, 32);
        if (ov > bv || (ov == bv && oj < bj)) { bv = ov; bj = oj; }
      }
      summax += bv;
      row[bj] = -__builtin_inff();
    }

    // nnPU combine: pos = 0.5*d_ap ; neg = -((1-0.5)/(1*40))*sum
    const float pos = 0.5f * dist_ap;
    const float neg = -0.0125f * (summin + summax);
    const float li  = (neg < 0.0f) ? (-neg) : (pos + neg);   // GAMMA = 1
    if (lane == 0) lossOut[g] = li;
  }
}

// ---------------------------------------------------------------------------
// Kernel 3: deterministic mean over 4096 losses -> d_out[0]
// ---------------------------------------------------------------------------
__global__ void tnnpu_reduce(const float* __restrict__ loss, float* __restrict__ out) {
  __shared__ float sbuf[256];
  float s = 0.0f;
  for (int i = threadIdx.x; i < NROWS; i += 256) s += loss[i];
  sbuf[threadIdx.x] = s;
  __syncthreads();
  for (int st = 128; st > 0; st >>= 1) {
    if ((int)threadIdx.x < st) sbuf[threadIdx.x] += sbuf[threadIdx.x + st];
    __syncthreads();
  }
  if (threadIdx.x == 0) out[0] = sbuf[0] * (1.0f / (float)NROWS);
}

// ---------------------------------------------------------------------------
extern "C" void kernel_launch(void* const* d_in, const int* in_sizes, int n_in,
                              void* d_out, int out_size, void* d_ws, size_t ws_size,
                              hipStream_t stream) {
  (void)in_sizes; (void)n_in; (void)out_size; (void)ws_size;

  const float* inp = (const float*)d_in[0];
  const float* tgt = (const float*)d_in[1];

  char* ws = (char*)d_ws;
  const size_t f16MatBytes = (size_t)NROWS * CDIM * sizeof(_Float16);   // 2 MB
  _Float16* Af   = (_Float16*)(ws);
  _Float16* Bf   = (_Float16*)(ws + f16MatBytes);
  float*    loss = (float*)(ws + 2 * f16MatBytes);
  float*    out  = (float*)d_out;

  const int smemBytes = 16 * STRIDEF * (int)sizeof(float);   // ~262 KB (< 320 KB/WGP)
  (void)hipFuncSetAttribute((const void*)tnnpu_gemm_topk,
                            hipFuncAttributeMaxDynamicSharedMemorySize, smemBytes);

  tnnpu_normalize<<<1024, 256, 0, stream>>>(inp, tgt, Af, Bf);
  tnnpu_gemm_topk<<<256, 256, smemBytes, stream>>>(Af, Bf, loss);
  tnnpu_reduce<<<1, 256, 0, stream>>>(loss, out);
}
